// PoolActorV2_7310034338363
// MI455X (gfx1250) — matrix operand order repaired and checked by reading, not compiled, not run on hardware
//
#include <hip/hip_runtime.h>

#define B_N      128
#define M_N      200
#define S_N      20
#define H_N      3
#define C_N      100
#define L_N      5
#define HID_N    150
#define E_N      3200
#define K_TOP    160
#define NEDGE    (E_N + M_N)          // 3400 (edges + self loops)
#define OBS_DIM  9002                 // 5*M+2 + 2*M*S
#define D2_IN    49600                // 100 + 3*160*100 + 2*5*150
#define NEG_GAT  0.2f
#define NEG_OUT  0.01f
#define NEGINF  (-3.402823e38f)

typedef __attribute__((ext_vector_type(16))) _Float16 v16h;
typedef __attribute__((ext_vector_type(8)))  _Float16 v8h;
typedef __attribute__((ext_vector_type(8)))  float    v8f;

// ---------------------------------------------------------------------------
// CDNA5 V_WMMA_F32_16X16X32_F16 fragment layouts (wave32):
// A frag: lane = row (0..15 twice); halves 0-7 hold K=(lane<16?0:8)+0..7,
//         halves 8-15 hold K=(lane<16?16:24)+0..7   -> two aligned v8h loads
// B frag: lane covers (K-half, N): halves j hold K=(lane<16?0:8)+(j&7)+(j&8?16:0), N=lane&15
//         -> pre-packed so each lane's 16 halves are 32 contiguous bytes
// C/D:    VGPR r: lanes0-15 -> (M=r, N=lane), lanes16-31 -> (M=8+r, N=lane-16)
// ---------------------------------------------------------------------------
__device__ __forceinline__ v16h load_a_frag(const _Float16* __restrict__ ap,
                                            int k0, int lane) {
  int kh = (lane < 16) ? 0 : 8;
  v8h lo = *(const v8h*)(ap + k0 + kh);
  v8h hi = *(const v8h*)(ap + k0 + 16 + kh);
  return __builtin_shufflevector(lo, hi, 0, 1, 2, 3, 4, 5, 6, 7,
                                 8, 9, 10, 11, 12, 13, 14, 15);
}

__device__ __forceinline__ v8f wmma_f16(v16h a, v16h b, v8f c) {
  return __builtin_amdgcn_wmma_f32_16x16x32_f16(false, a, false, b,
                                                (short)0, c, false, false);
}

// Pack f32 (rows x cols) weight matrix into WMMA B-fragment tiles:
// dst[((kt*ntiles + nt)*32 + lane)*16 + j] = src[K, N] (zero-padded)
// with K = kt*32 + ((lane&16)>>1) + (j&7) + ((j&8)<<1), N = nt*16 + (lane&15)
__global__ void pack_b_frag(const float* __restrict__ src, _Float16* __restrict__ dst,
                            int rows, int cols, int ktiles, int ntiles) {
  int idx = blockIdx.x * blockDim.x + threadIdx.x;
  if (idx >= ktiles * ntiles * 512) return;
  int j    = idx & 15;
  int lane = (idx >> 4) & 31;
  int nt   = (idx >> 9) % ntiles;
  int kt   = idx / (512 * ntiles);
  int K    = kt * 32 + ((lane & 16) >> 1) + (j & 7) + ((j & 8) << 1);
  int N    = nt * 16 + (lane & 15);
  float v  = (K < rows && N < cols) ? src[(size_t)K * cols + N] : 0.f;
  dst[idx] = (_Float16)v;
}

// Generic C(MxN) = A(MxK)f16(row major) * Bp(packed frags) (+bias).
// One wave per 16x16 tile; K % 32 == 0; A rows padded to 16*gridDim.y.
__global__ void gemm16(const _Float16* __restrict__ A,
                       const _Float16* __restrict__ Bp,
                       float* __restrict__ C,
                       int K, int lda, int ntiles, int ldc,
                       int Mv, int Nv, const float* __restrict__ bias) {
  int lane = threadIdx.x & 31;
  int nt   = blockIdx.x;
  int row0 = blockIdx.y * 16;
  const _Float16* ap = A + (size_t)(row0 + (lane & 15)) * lda;
  const v16h* bt = (const v16h*)Bp + (size_t)nt * 32 + lane;
  const int bstep = ntiles * 32;
  v8f acc = {};
  for (int k0 = 0; k0 < K; k0 += 32) {
    __builtin_prefetch(ap + k0 + 64, 0, 1);          // global_prefetch_b8
    __builtin_prefetch(bt + bstep, 0, 1);
    v16h a = load_a_frag(ap, k0, lane);
    v16h b = *bt;                                    // 2x global_load_b128
    bt += bstep;
    acc = wmma_f16(a, b, acc);
  }
  int nn = lane & 15;
  int mb = (lane < 16) ? 0 : 8;
  int col = nt * 16 + nn;
  if (col >= Nv) return;
  float bv = bias ? bias[col] : 0.f;
#pragma unroll
  for (int i = 0; i < 8; i++) {
    int row = row0 + mb + i;
    if (row < Mv) C[(size_t)row * ldc + col] = acc[i] + bv;
  }
}

// ---------------------------------------------------------------------------
// LSTM: gates = x_t @ Wih + h_{t-1} @ Whh  (both dirs; fused dual-K WMMA)
// xseq (L,B,288)f16; wihp/whhp packed frags per dir; hprev (2,128,160)f16
// ---------------------------------------------------------------------------
__global__ void lstm_gates(const _Float16* __restrict__ xseq,
                           const _Float16* __restrict__ wihp,
                           const _Float16* __restrict__ whhp,
                           const _Float16* __restrict__ hprev,
                           float* __restrict__ gates, int t) {
  int lane = threadIdx.x & 31;
  int nt   = blockIdx.x;                 // 0..37
  int row0 = blockIdx.y * 16;
  int dir  = blockIdx.z;
  int time = (dir == 0) ? t : (L_N - 1 - t);
  const _Float16* ax = xseq + (size_t)time * B_N * 288
                            + (size_t)(row0 + (lane & 15)) * 288;
  const _Float16* ah = hprev + (size_t)dir * B_N * 160
                             + (size_t)(row0 + (lane & 15)) * 160;
  const v16h* bw = (const v16h*)(wihp + (size_t)dir * 9 * 38 * 512)
                   + (size_t)nt * 32 + lane;
  const v16h* bh = (const v16h*)(whhp + (size_t)dir * 5 * 38 * 512)
                   + (size_t)nt * 32 + lane;
  float* G = gates + (size_t)dir * B_N * 608;

  v8f acc = {};
#pragma unroll 3
  for (int kt = 0; kt < 9; kt++) {
    v16h a = load_a_frag(ax, kt * 32, lane);
    v16h b = *bw; bw += 38 * 32;
    acc = wmma_f16(a, b, acc);
  }
#pragma unroll
  for (int kt = 0; kt < 5; kt++) {
    v16h a = load_a_frag(ah, kt * 32, lane);
    v16h b = *bh; bh += 38 * 32;
    acc = wmma_f16(a, b, acc);
  }
  int nn = lane & 15;
  int mb = (lane < 16) ? 0 : 8;
#pragma unroll
  for (int i = 0; i < 8; i++)
    G[(row0 + mb + i) * 608 + nt * 16 + nn] = acc[i];
}

__device__ __forceinline__ float sigmoidf(float x) {
  return 1.f / (1.f + expf(-x));
}

__global__ void lstm_pointwise(const float* __restrict__ gates,
                               float* __restrict__ cst,
                               _Float16* __restrict__ hh,
                               _Float16* __restrict__ hidden_in,
                               const float* __restrict__ bih_f,
                               const float* __restrict__ bhh_f,
                               const float* __restrict__ bih_b,
                               const float* __restrict__ bhh_b, int t) {
  int idx = blockIdx.x * blockDim.x + threadIdx.x;
  if (idx >= 2 * B_N * HID_N) return;
  int j   = idx % HID_N;
  int b   = (idx / HID_N) % B_N;
  int dir = idx / (HID_N * B_N);
  const float* G  = gates + (size_t)dir * B_N * 608 + (size_t)b * 608;
  const float* bi = dir ? bih_b : bih_f;
  const float* bh = dir ? bhh_b : bhh_f;
  float gi = G[j]       + bi[j]       + bh[j];
  float gf = G[150 + j] + bi[150 + j] + bh[150 + j];
  float gg = G[300 + j] + bi[300 + j] + bh[300 + j];
  float go = G[450 + j] + bi[450 + j] + bh[450 + j];
  float* cp = cst + (size_t)dir * B_N * HID_N + (size_t)b * HID_N + j;
  float c = sigmoidf(gf) * (*cp) + sigmoidf(gi) * tanhf(gg);
  float h = sigmoidf(go) * tanhf(c);
  *cp = c;
  hh[(size_t)dir * B_N * 160 + (size_t)b * 160 + j] = (_Float16)h;
  int time = (dir == 0) ? t : (L_N - 1 - t);
  hidden_in[(size_t)b * D2_IN + 48100 + time * 300 + dir * 150 + j] = (_Float16)h;
}

// ---------------------------------------------------------------------------
// Elementwise / conversion kernels
// ---------------------------------------------------------------------------
__global__ void fill_f32(float* p, int n, float v) {
  int i = blockIdx.x * blockDim.x + threadIdx.x;
  if (i < n) p[i] = v;
}
__global__ void fill_f16(_Float16* p, int n) {
  int i = blockIdx.x * blockDim.x + threadIdx.x;
  if (i < n) p[i] = (_Float16)0.f;
}

__global__ void build_xseq(const float* __restrict__ past, _Float16* __restrict__ xseq) {
  int idx = blockIdx.x * blockDim.x + threadIdx.x;
  if (idx >= L_N * B_N * 288) return;
  int j = idx % 288, b = (idx / 288) % B_N, t = idx / (288 * B_N);
  xseq[idx] = (_Float16)((j < 277) ? past[((size_t)b * L_N + t) * 277 + j] : 0.f);
}

__global__ void build_sstate(const float* __restrict__ obs, _Float16* __restrict__ s) {
  int idx = blockIdx.x * blockDim.x + threadIdx.x;
  if (idx >= B_N * 608) return;
  int j = idx % 608, b = idx / 608;
  s[idx] = (_Float16)((j < 602) ? obs[(size_t)b * OBS_DIM + j] : 0.f);
}

__global__ void build_feat(const float* __restrict__ obs, _Float16* __restrict__ feat) {
  int idx = blockIdx.x * blockDim.x + threadIdx.x;
  if (idx >= B_N * M_N * 64) return;
  int col = idx & 63;
  int m   = (idx >> 6) % M_N;
  int b   = idx / (64 * M_N);
  const float* o = obs + (size_t)b * OBS_DIM;
  float v = 0.f;
  if (col == 0)       v = o[3 * M_N + 2 + m];
  else if (col == 1)  v = o[4 * M_N + 2 + m];
  else if (col < 22)  v = o[5 * M_N + 2 + m * S_N + (col - 2)];
  else if (col < 42)  v = o[5 * M_N + 2 + M_N * S_N + m * S_N + (col - 22)];
  feat[idx] = (_Float16)v;
}

// ---------------------------------------------------------------------------
// GAT edge kernels (segment softmax via atomics)
// ---------------------------------------------------------------------------
__device__ __forceinline__ void atomicMaxF(float* addr, float val) {
  unsigned int* ua = (unsigned int*)addr;
  unsigned int old = *ua;
  while (true) {
    float f = __uint_as_float(old);
    if (f >= val) break;
    unsigned int assumed = old;
    old = atomicCAS(ua, assumed, __float_as_uint(val));
    if (old == assumed) break;
  }
}

__device__ __forceinline__ void edge_st(const int* ei, int e, int& src, int& tgt) {
  if (e < E_N) { src = ei[e]; tgt = ei[E_N + e]; }
  else         { src = e - E_N; tgt = e - E_N; }
}

__global__ void edge_logits(const float* __restrict__ xl, const float* __restrict__ xr,
                            const int* __restrict__ ei, const float* __restrict__ att,
                            float* __restrict__ logits, float* __restrict__ mx) {
  int idx = blockIdx.x * blockDim.x + threadIdx.x;
  if (idx >= B_N * NEDGE * H_N) return;
  int h = idx % H_N, e = (idx / H_N) % NEDGE, b = idx / (H_N * NEDGE);
  int src, tgt; edge_st(ei, e, src, tgt);
  const float* pl = xl + ((size_t)(b * M_N + src) * 300 + h * 100);
  const float* pr = xr + ((size_t)(b * M_N + tgt) * 300 + h * 100);
  const float* pa = att + h * 100;
  float s = 0.f;
  for (int c = 0; c < 100; c++) {
    float v = pl[c] + pr[c];
    v = (v > 0.f) ? v : NEG_GAT * v;
    s += v * pa[c];
  }
  logits[idx] = s;
  atomicMaxF(&mx[(size_t)(b * M_N + tgt) * H_N + h], s);
}

__global__ void edge_softmax(float* __restrict__ logits, const float* __restrict__ mx,
                             float* __restrict__ den, const int* __restrict__ ei) {
  int idx = blockIdx.x * blockDim.x + threadIdx.x;
  if (idx >= B_N * NEDGE * H_N) return;
  int h = idx % H_N, e = (idx / H_N) % NEDGE, b = idx / (H_N * NEDGE);
  int src, tgt; edge_st(ei, e, src, tgt); (void)src;
  size_t ti = (size_t)(b * M_N + tgt) * H_N + h;
  float ex = expf(logits[idx] - mx[ti]);
  logits[idx] = ex;                     // reuse buffer for ex
  atomicAdd(&den[ti], ex);
}

__global__ void edge_agg(const float* __restrict__ ex, const float* __restrict__ den,
                         const float* __restrict__ xl, const int* __restrict__ ei,
                         float* __restrict__ agg) {
  int idx = blockIdx.x * blockDim.x + threadIdx.x;
  if (idx >= B_N * NEDGE * H_N) return;
  int h = idx % H_N, e = (idx / H_N) % NEDGE, b = idx / (H_N * NEDGE);
  int src, tgt; edge_st(ei, e, src, tgt);
  size_t ti = (size_t)(b * M_N + tgt) * H_N + h;
  float alpha = ex[idx] / den[ti];
  const float* pl = xl + ((size_t)(b * M_N + src) * 300 + h * 100);
  float* pa = agg + ((size_t)(b * M_N + tgt) * 300 + h * 100);
  for (int c = 0; c < 100; c++) atomicAdd(&pa[c], pl[c] * alpha);
}

__global__ void add_gat_bias(float* __restrict__ agg, const float* __restrict__ bias) {
  int idx = blockIdx.x * blockDim.x + threadIdx.x;
  if (idx >= B_N * M_N * 300) return;
  agg[idx] += bias[idx % 300];
}

// ---------------------------------------------------------------------------
// Pooling
// ---------------------------------------------------------------------------
__global__ void pool_norm_kernel(const float* __restrict__ pw, float* __restrict__ invn) {
  __shared__ float red[256];
  int tid = threadIdx.x;
  float s = 0.f;
  for (int i = tid; i < 300; i += 256) { float v = pw[i]; s += v * v; }
  red[tid] = s; __syncthreads();
  for (int off = 128; off > 0; off >>= 1) {
    if (tid < off) red[tid] += red[tid + off];
    __syncthreads();
  }
  if (tid == 0) invn[0] = 1.f / sqrtf(red[0]);
}

__global__ void score_kernel(const float* __restrict__ gat_out,
                             const float* __restrict__ pw,
                             const float* __restrict__ invn,
                             float* __restrict__ score) {
  int wid  = (blockIdx.x * blockDim.x + threadIdx.x) >> 5;
  int lane = threadIdx.x & 31;
  if (wid >= B_N * M_N) return;
  const float* g = gat_out + (size_t)wid * 300;
  float s = 0.f;
  for (int c = lane; c < 300; c += 32) s += g[c] * pw[c];
#pragma unroll
  for (int off = 16; off > 0; off >>= 1) s += __shfl_xor(s, off, 32);
  if (lane == 0) score[wid] = tanhf(s * invn[0]);
}

__global__ void topk_kernel(const float* __restrict__ score,
                            int* __restrict__ topi, float* __restrict__ topv) {
  __shared__ float s[M_N];
  __shared__ float bv[256];
  __shared__ int   bi[256];
  int b = blockIdx.x, tid = threadIdx.x;
  for (int i = tid; i < M_N; i += blockDim.x) s[i] = score[b * M_N + i];
  __syncthreads();
  for (int r = 0; r < K_TOP; r++) {
    float best = NEGINF; int besti = -1;
    for (int i = tid; i < M_N; i += blockDim.x) {
      float v = s[i];
      if (v > best || (v == best && (unsigned)i < (unsigned)besti)) { best = v; besti = i; }
    }
    bv[tid] = best; bi[tid] = besti;
    __syncthreads();
    for (int off = 128; off > 0; off >>= 1) {
      if (tid < off) {
        if (bv[tid + off] > bv[tid] ||
            (bv[tid + off] == bv[tid] && (unsigned)bi[tid + off] < (unsigned)bi[tid])) {
          bv[tid] = bv[tid + off]; bi[tid] = bi[tid + off];
        }
      }
      __syncthreads();
    }
    if (tid == 0) {
      topi[b * K_TOP + r] = bi[0];
      topv[b * K_TOP + r] = bv[0];
      s[bi[0]] = NEGINF;
    }
    __syncthreads();
  }
}

__global__ void pool_gather(const float* __restrict__ agg, const int* __restrict__ topi,
                            const float* __restrict__ topv, _Float16* __restrict__ hin) {
  int idx = blockIdx.x * blockDim.x + threadIdx.x;
  if (idx >= B_N * K_TOP * 300) return;
  int c = idx % 300, k = (idx / 300) % K_TOP, b = idx / (300 * K_TOP);
  int m = topi[b * K_TOP + k];
  float v = agg[((size_t)b * M_N + m) * 300 + c] * topv[b * K_TOP + k];
  hin[(size_t)b * D2_IN + 100 + k * 300 + c] = (_Float16)v;
}

// ---------------------------------------------------------------------------
// Dense epilogues
// ---------------------------------------------------------------------------
__global__ void d1_elu(const float* __restrict__ d1out, _Float16* __restrict__ hin) {
  int idx = blockIdx.x * blockDim.x + threadIdx.x;
  if (idx >= B_N * 100) return;
  int j = idx % 100, b = idx / 100;
  float x = d1out[b * 100 + j];
  x = (x > 0.f) ? x : (expf(x) - 1.f);
  hin[(size_t)b * D2_IN + j] = (_Float16)x;
}

__global__ void d2_elu(const float* __restrict__ hid, _Float16* __restrict__ hid_h) {
  int idx = blockIdx.x * blockDim.x + threadIdx.x;
  if (idx >= B_N * 128) return;
  float x = hid[idx];
  x = (x > 0.f) ? x : (expf(x) - 1.f);
  hid_h[idx] = (_Float16)x;
}

__global__ void out_leaky(float* __restrict__ out) {
  int idx = blockIdx.x * blockDim.x + threadIdx.x;
  if (idx >= B_N * M_N) return;
  float v = out[idx];
  out[idx] = (v > 0.f) ? v : NEG_OUT * v;
}

// ---------------------------------------------------------------------------
// Host driver
// ---------------------------------------------------------------------------
extern "C" void kernel_launch(void* const* d_in, const int* in_sizes, int n_in,
                              void* d_out, int out_size, void* d_ws, size_t ws_size,
                              hipStream_t stream) {
  (void)in_sizes; (void)n_in; (void)out_size; (void)ws_size;
  const float* obs    = (const float*)d_in[0];
  const float* past   = (const float*)d_in[1];
  const int*   ei     = (const int*)d_in[2];
  const float* d1_w   = (const float*)d_in[3];
  const float* d1_b   = (const float*)d_in[4];
  const float* d2_w   = (const float*)d_in[5];
  const float* d2_b   = (const float*)d_in[6];
  const float* gat_wl = (const float*)d_in[7];
  const float* gat_bl = (const float*)d_in[8];
  const float* gat_wr = (const float*)d_in[9];
  const float* gat_br = (const float*)d_in[10];
  const float* gat_att= (const float*)d_in[11];
  const float* gat_bias=(const float*)d_in[12];
  const float* pool_w = (const float*)d_in[13];
  const float* wih_f  = (const float*)d_in[14];
  const float* whh_f  = (const float*)d_in[15];
  const float* bih_f  = (const float*)d_in[16];
  const float* bhh_f  = (const float*)d_in[17];
  const float* wih_b  = (const float*)d_in[18];
  const float* whh_b  = (const float*)d_in[19];
  const float* bih_b  = (const float*)d_in[20];
  const float* bhh_b  = (const float*)d_in[21];
  const float* out_w  = (const float*)d_in[22];
  const float* out_b  = (const float*)d_in[23];
  float* out = (float*)d_out;

  char* wp = (char*)d_ws;
  auto alloc = [&](size_t n) -> char* {
    char* r = wp; wp += (n + 255) & ~(size_t)255; return r;
  };
  // packed B-fragment buffers (ktiles*ntiles*512 halves each)
  _Float16* d2w_p   = (_Float16*)alloc(sizeof(_Float16) * (size_t)1550 * 8 * 512);
  _Float16* gatwl_p = (_Float16*)alloc(sizeof(_Float16) * 2 * 19 * 512);
  _Float16* gatwr_p = (_Float16*)alloc(sizeof(_Float16) * 2 * 19 * 512);
  _Float16* d1w_p   = (_Float16*)alloc(sizeof(_Float16) * 19 * 7 * 512);
  _Float16* outw_p  = (_Float16*)alloc(sizeof(_Float16) * 4 * 13 * 512);
  _Float16* wih_p   = (_Float16*)alloc(sizeof(_Float16) * 2 * 9 * 38 * 512);
  _Float16* whh_p   = (_Float16*)alloc(sizeof(_Float16) * 2 * 5 * 38 * 512);
  // activations
  _Float16* hin_h   = (_Float16*)alloc(sizeof(_Float16) * (size_t)B_N * D2_IN);
  _Float16* feat_h  = (_Float16*)alloc(sizeof(_Float16) * (size_t)B_N * M_N * 64);
  float*    xl      = (float*)alloc(sizeof(float) * (size_t)B_N * M_N * 300);
  float*    xr      = (float*)alloc(sizeof(float) * (size_t)B_N * M_N * 300);
  float*    logits  = (float*)alloc(sizeof(float) * (size_t)B_N * NEDGE * H_N);
  float*    mx      = (float*)alloc(sizeof(float) * (size_t)B_N * M_N * H_N);
  float*    den     = (float*)alloc(sizeof(float) * (size_t)B_N * M_N * H_N);
  float*    agg     = (float*)alloc(sizeof(float) * (size_t)B_N * M_N * 300);
  float*    score   = (float*)alloc(sizeof(float) * B_N * M_N);
  int*      topi    = (int*)alloc(sizeof(int) * B_N * K_TOP);
  float*    topv    = (float*)alloc(sizeof(float) * B_N * K_TOP);
  float*    invn    = (float*)alloc(sizeof(float));
  _Float16* xseq_h  = (_Float16*)alloc(sizeof(_Float16) * L_N * B_N * 288);
  float*    gates   = (float*)alloc(sizeof(float) * 2 * B_N * 608);
  float*    cstate  = (float*)alloc(sizeof(float) * 2 * B_N * HID_N);
  _Float16* hprev_h = (_Float16*)alloc(sizeof(_Float16) * 2 * B_N * 160);
  _Float16* sst_h   = (_Float16*)alloc(sizeof(_Float16) * B_N * 608);
  float*    d1out   = (float*)alloc(sizeof(float) * B_N * 100);
  float*    hid     = (float*)alloc(sizeof(float) * B_N * 128);
  _Float16* hid_h   = (_Float16*)alloc(sizeof(_Float16) * B_N * 128);

  const int T = 256;
  auto nb = [](long n) { return (int)((n + 255) / 256); };

  // --- weight packing into WMMA B-fragment layout (f32 -> f16, zero-padded) ---
  pack_b_frag<<<nb((long)1550 * 8 * 512), T, 0, stream>>>(d2_w, d2w_p, D2_IN, 128, 1550, 8);
  pack_b_frag<<<nb(2 * 19 * 512), T, 0, stream>>>(gat_wl, gatwl_p, 42, 300, 2, 19);
  pack_b_frag<<<nb(2 * 19 * 512), T, 0, stream>>>(gat_wr, gatwr_p, 42, 300, 2, 19);
  pack_b_frag<<<nb(19 * 7 * 512), T, 0, stream>>>(d1_w, d1w_p, 602, 100, 19, 7);
  pack_b_frag<<<nb(4 * 13 * 512), T, 0, stream>>>(out_w, outw_p, 128, 200, 4, 13);
  pack_b_frag<<<nb(9 * 38 * 512), T, 0, stream>>>(wih_f, wih_p, 277, 600, 9, 38);
  pack_b_frag<<<nb(9 * 38 * 512), T, 0, stream>>>(wih_b, wih_p + 9 * 38 * 512, 277, 600, 9, 38);
  pack_b_frag<<<nb(5 * 38 * 512), T, 0, stream>>>(whh_f, whh_p, 150, 600, 5, 38);
  pack_b_frag<<<nb(5 * 38 * 512), T, 0, stream>>>(whh_b, whh_p + 5 * 38 * 512, 150, 600, 5, 38);
  build_xseq<<<nb(L_N * B_N * 288), T, 0, stream>>>(past, xseq_h);
  build_sstate<<<nb(B_N * 608), T, 0, stream>>>(obs, sst_h);
  build_feat<<<nb((long)B_N * M_N * 64), T, 0, stream>>>(obs, feat_h);

  // --- state init ---
  fill_f32<<<nb(2 * B_N * HID_N), T, 0, stream>>>(cstate, 2 * B_N * HID_N, 0.f);
  fill_f16<<<nb(2 * B_N * 160), T, 0, stream>>>(hprev_h, 2 * B_N * 160);
  fill_f32<<<nb(B_N * M_N * H_N), T, 0, stream>>>(mx, B_N * M_N * H_N, NEGINF);
  fill_f32<<<nb(B_N * M_N * H_N), T, 0, stream>>>(den, B_N * M_N * H_N, 0.f);
  fill_f32<<<nb((long)B_N * M_N * 300), T, 0, stream>>>(agg, B_N * M_N * 300, 0.f);

  // --- bidirectional LSTM (5 sequential steps, both dirs per launch) ---
  for (int t = 0; t < L_N; t++) {
    lstm_gates<<<dim3(38, 8, 2), 32, 0, stream>>>(xseq_h, wih_p, whh_p, hprev_h, gates, t);
    lstm_pointwise<<<nb(2 * B_N * HID_N), T, 0, stream>>>(
        gates, cstate, hprev_h, hin_h, bih_f, bhh_f, bih_b, bhh_b, t);
  }

  // --- GAT projections: (B*M, 42p64) @ (64, 300p304) ---
  gemm16<<<dim3(19, (B_N * M_N) / 16), 32, 0, stream>>>(
      feat_h, gatwl_p, xl, 64, 64, 19, 300, B_N * M_N, 300, gat_bl);
  gemm16<<<dim3(19, (B_N * M_N) / 16), 32, 0, stream>>>(
      feat_h, gatwr_p, xr, 64, 64, 19, 300, B_N * M_N, 300, gat_br);

  // --- GAT edge softmax + aggregation ---
  long ne3 = (long)B_N * NEDGE * H_N;
  edge_logits<<<nb(ne3), T, 0, stream>>>(xl, xr, ei, gat_att, logits, mx);
  edge_softmax<<<nb(ne3), T, 0, stream>>>(logits, mx, den, ei);
  edge_agg<<<nb(ne3), T, 0, stream>>>(logits, den, xl, ei, agg);
  add_gat_bias<<<nb((long)B_N * M_N * 300), T, 0, stream>>>(agg, gat_bias);

  // --- score / top-k pooling into d2 input ---
  pool_norm_kernel<<<1, 256, 0, stream>>>(pool_w, invn);
  score_kernel<<<nb((long)B_N * M_N * 32), T, 0, stream>>>(agg, pool_w, invn, score);
  topk_kernel<<<B_N, 256, 0, stream>>>(score, topi, topv);
  pool_gather<<<nb((long)B_N * K_TOP * 300), T, 0, stream>>>(agg, topi, topv, hin_h);

  // --- dense layers ---
  gemm16<<<dim3(7, 8), 32, 0, stream>>>(sst_h, d1w_p, d1out, 608, 608, 7, 100,
                                        B_N, 100, d1_b);
  d1_elu<<<nb(B_N * 100), T, 0, stream>>>(d1out, hin_h);

  gemm16<<<dim3(8, 8), 32, 0, stream>>>(hin_h, d2w_p, hid, D2_IN, D2_IN, 8, 128,
                                        B_N, 128, d2_b);
  d2_elu<<<nb(B_N * 128), T, 0, stream>>>(hid, hid_h);

  gemm16<<<dim3(13, 8), 32, 0, stream>>>(hid_h, outw_p, out, 128, 128, 13, 200,
                                         B_N, 200, out_b);
  out_leaky<<<nb(B_N * M_N), T, 0, stream>>>(out);
}
